// Gat_Fusion_16930761081284
// MI455X (gfx1250) — matrix operand order repaired and checked by reading, not compiled, not run on hardware
//
#include <hip/hip_runtime.h>
#include <hip/hip_bf16.h>

typedef __attribute__((ext_vector_type(16))) __bf16 v16bf;
typedef __attribute__((ext_vector_type(8)))  __bf16 v8bf;
typedef __attribute__((ext_vector_type(4)))  __bf16 v4bf;
typedef __attribute__((ext_vector_type(8)))  float  v8f;

#define GB     8      // batches per workgroup
#define NODES  12
#define DIM    256
#define WELEM  (DIM * DIM)     // elements per weight matrix
#define SROW   264    // LDS row stride in bf16 elems (256 + 8 pad; 528B, 16B aligned)
#define NEGC   9e15f

// ================= pre-pass: convert 5 weight matrices fp32 -> bf16 in d_ws ==========
__global__ __launch_bounds__(256)
void cvt_weights_kernel(const float* __restrict__ Wv, const float* __restrict__ Wk,
                        const float* __restrict__ Wq, const float* __restrict__ Wo1,
                        const float* __restrict__ Wf, __bf16* __restrict__ ws) {
    int t = blockIdx.x * 256 + threadIdx.x;       // 0 .. 5*WELEM/8 - 1
    int w = t / (WELEM / 8);
    int e = (t % (WELEM / 8)) * 8;
    const float* srcs[5] = { Wv, Wk, Wq, Wo1, Wf };
    const float4* p = (const float4*)(srcs[w] + e);
    float4 a = p[0], b = p[1];
    v8bf o;
    o[0]=(__bf16)a.x; o[1]=(__bf16)a.y; o[2]=(__bf16)a.z; o[3]=(__bf16)a.w;
    o[4]=(__bf16)b.x; o[5]=(__bf16)b.y; o[6]=(__bf16)b.z; o[7]=(__bf16)b.w;
    *(v8bf*)(ws + (size_t)w * WELEM + e) = o;
}

// ---- fragment helpers (layouts per CDNA5 ISA §7.12.2) ----

// 16-bit A 16x32: lane m (m<16): K 0..7 & 16..23 ; lane m+16: K 8..15 & 24..31
__device__ __forceinline__ v16bf load_a_frag(const __bf16* rowptr, int kbase, int lane) {
    int hi = (lane >> 4) & 1;
    v8bf a0 = *(const v8bf*)(rowptr + kbase + hi * 8);
    v8bf a1 = *(const v8bf*)(rowptr + kbase + 16 + hi * 8);
    v16bf a;
#pragma unroll
    for (int i = 0; i < 8; ++i) { a[i] = a0[i]; a[i + 8] = a1[i]; }
    return a;
}

// 16-bit B 32x16 (B[k][n] = W[n][k]): lane n<16 holds col n, K 0..7 & 16..23; lane n+16: K 8..15 & 24..31.
// W is pre-converted bf16 row-major [256][256] -> two runs of 8 contiguous bf16 per lane, 16 apart (2x b128).
__device__ __forceinline__ v16bf load_b_frag(const __bf16* W, int ntile, int kstep, int lane) {
    int col = ntile * 16 + (lane & 15);
    int k0  = kstep * 32 + ((lane >> 4) & 1) * 8;
    const v8bf* p = (const v8bf*)(W + (size_t)col * DIM + k0);
    v8bf b0 = p[0], b1 = p[2];
    v16bf b;
#pragma unroll
    for (int i = 0; i < 8; ++i) { b[i] = b0[i]; b[i + 8] = b1[i]; }
    return b;
}

// 32-bit C/D 16x16: VGPR v -> M = v (lanes<16) or v+8 (lanes>=16); N = lane&15
__device__ __forceinline__ void store_c_lds(int row0, __bf16* buf, const float* bias,
                                            int ntile, v8f acc, int lane, bool relu) {
    int n  = lane & 15;
    int mb = ((lane >> 4) & 1) * 8;
    float bb = bias[ntile * 16 + n];
#pragma unroll
    for (int v = 0; v < 8; ++v) {
        float x = acc[v] + bb;
        if (relu) x = fmaxf(x, 0.0f);
        buf[(size_t)(row0 + v + mb) * SROW + ntile * 16 + n] = (__bf16)x;
    }
}

// One output column tile (16 cols) of a GEMM, MT M-tiles deep.
// B fragment loaded ONCE per k-step and reused across all MT WMMAs.
template<int MT, bool RELU>
__device__ __forceinline__ void gemm_col(const __bf16* __restrict__ Wbf,
                                         const float* __restrict__ bias,
                                         __bf16* __restrict__ dst,
                                         const __bf16* const* arows,
                                         int ntile, int lane) {
    v8f acc[MT];
#pragma unroll
    for (int m = 0; m < MT; ++m) { v8f z = {}; acc[m] = z; }
#pragma unroll
    for (int ks = 0; ks < 8; ++ks) {
        v16bf bfrag = load_b_frag(Wbf, ntile, ks, lane);
#pragma unroll
        for (int m = 0; m < MT; ++m) {
            v16bf afrag = load_a_frag(arows[m], ks * 32, lane);
            acc[m] = __builtin_amdgcn_wmma_f32_16x16x32_bf16(false, afrag, false, bfrag,
                                                             (short)0, acc[m], false, false);
        }
    }
#pragma unroll
    for (int m = 0; m < MT; ++m)
        store_c_lds(m * 16, dst, bias, ntile, acc[m], lane, RELU);
}

__global__ __launch_bounds__(256, 1)
void gat_fused_kernel(const float* __restrict__ h,   const float* __restrict__ adj,
                      const float* __restrict__ lab,
                      const __bf16* __restrict__ Wbf,            // 5 matrices in d_ws
                      const float* __restrict__ pbv, const float* __restrict__ pbk,
                      const float* __restrict__ pbq, const float* __restrict__ pbo1,
                      const float* __restrict__ pbf,
                      float* __restrict__ out) {
    __shared__ __bf16 sh_h [GB * NODES * SROW];   // staged input (bf16)
    __shared__ __bf16 sh_v [GB * NODES * SROW];   // relu(h Wv^T + bv)
    __shared__ __bf16 sh_k [GB * NODES * SROW];   // relu(h Wk^T + bk)
    __shared__ __bf16 sh_q [GB * 4 * SROW];       // relu(h_sel Wq^T + bq)
    __shared__ __bf16 sh_f1[GB * 4 * SROW];       // att @ Wh1
    __shared__ __bf16 sh_f2[GB * 4 * SROW];       // f1 Wo1^T + bo1
    __shared__ float  sh_sc [GB * 4 * NODES];     // masked scores
    __shared__ float  sh_att[GB * 4 * NODES];     // softmax * label

    const int tid  = threadIdx.x;
    const int lane = tid & 31;
    const int wave = tid >> 5;
    const int bbase = blockIdx.x * GB;

    const __bf16* WvB  = Wbf + 0 * WELEM;
    const __bf16* WkB  = Wbf + 1 * WELEM;
    const __bf16* WqB  = Wbf + 2 * WELEM;
    const __bf16* Wo1B = Wbf + 3 * WELEM;
    const __bf16* WfB  = Wbf + 4 * WELEM;

    __builtin_prefetch(WvB, 0, 1);  __builtin_prefetch(WkB, 0, 1);
    __builtin_prefetch(WqB, 0, 1);  __builtin_prefetch(Wo1B, 0, 1);
    __builtin_prefetch(WfB, 0, 1);

    // ---------- Phase 1: stage h -> LDS as bf16 ----------
    {
        const float4* hsrc = (const float4*)(h + (size_t)bbase * NODES * DIM);
        const int nvec = GB * NODES * DIM / 4;   // 6144 float4
#pragma unroll
        for (int it = 0; it < nvec / 256; ++it) {
            int fi = tid + it * 256;
            int r  = fi >> 6;            // / (DIM/4)
            int c  = (fi & 63) << 2;
            float4 f = hsrc[fi];
            v4bf o; o[0]=(__bf16)f.x; o[1]=(__bf16)f.y; o[2]=(__bf16)f.z; o[3]=(__bf16)f.w;
            *(v4bf*)(sh_h + (size_t)r * SROW + c) = o;
        }
    }
    __syncthreads();

    // ---------- Phase 2: projections. Job = (proj, ntile); B reused across M ----------
    for (int job = wave; job < 48; job += 8) {
        int proj = job >> 4, ntile = job & 15;
        if (proj < 2) {                               // V and K: all 96 rows, 6 M-tiles
            const __bf16* arows[6];
#pragma unroll
            for (int m = 0; m < 6; ++m)
                arows[m] = sh_h + (size_t)(m * 16 + (lane & 15)) * SROW;
            if (proj == 0) gemm_col<6, true>(WvB, pbv, sh_v, arows, ntile, lane);
            else           gemm_col<6, true>(WkB, pbk, sh_k, arows, ntile, lane);
        } else {                                      // Q: 32 selected rows, 2 M-tiles
            const __bf16* arows[2];
#pragma unroll
            for (int m = 0; m < 2; ++m) {
                int mrow = m * 16 + (lane & 15);      // = b_local*4 + i
                int b = mrow >> 2, i = mrow & 3;      // node = SEL[i] = 3*i
                arows[m] = sh_h + (size_t)(b * NODES + 3 * i) * SROW;
            }
            gemm_col<2, true>(WqB, pbq, sh_q, arows, ntile, lane);
        }
    }
    __syncthreads();

    // ---------- Phase 3: masked scores (only 4 selected query rows) ----------
    for (int e = tid; e < GB * 4 * NODES; e += 256) {
        int b = e / 48, rem = e % 48, i = rem / 12, j = rem % 12;
        const __bf16* qr = sh_q + (size_t)(b * 4 + i) * SROW;
        const __bf16* kr = sh_k + (size_t)(b * NODES + j) * SROW;
        float s = 0.f;
        for (int c = 0; c < DIM; c += 8) {
            v8bf qa = *(const v8bf*)(qr + c);
            v8bf ka = *(const v8bf*)(kr + c);
#pragma unroll
            for (int u = 0; u < 8; ++u) s += (float)qa[u] * (float)ka[u];
        }
        float av = adj[(3 * i) * NODES + j];
        sh_sc[e] = s * av - NEGC * (1.0f - av);
    }
    __syncthreads();

    // ---------- Phase 3b: softmax over j, times label ----------
    if (tid < GB * 4) {
        int b = tid >> 2, i = tid & 3;
        const float* sc = sh_sc + b * 48 + i * 12;
        float mx = sc[0];
#pragma unroll
        for (int j = 1; j < NODES; ++j) mx = fmaxf(mx, sc[j]);
        float ex[NODES], sum = 0.f;
#pragma unroll
        for (int j = 0; j < NODES; ++j) { ex[j] = __expf(sc[j] - mx); sum += ex[j]; }
        float inv = 1.0f / sum;
        const float* lb = lab + (size_t)(bbase + b) * NODES;
#pragma unroll
        for (int j = 0; j < NODES; ++j) sh_att[b * 48 + i * 12 + j] = ex[j] * inv * lb[j];
    }
    __syncthreads();

    // ---------- Phase 4: f1 = att @ Wh1 (32 rows x 256) ----------
    {
        int row = tid >> 3;               // 0..31
        int c0  = (tid & 7) * 32;
        int b = row >> 2, i = row & 3;
        float acc[32];
#pragma unroll
        for (int c = 0; c < 32; ++c) acc[c] = 0.f;
        for (int j = 0; j < NODES; ++j) {
            float a = sh_att[b * 48 + i * 12 + j];
            const __bf16* vr = sh_v + (size_t)(b * NODES + j) * SROW + c0;
#pragma unroll
            for (int c = 0; c < 32; ++c) acc[c] += a * (float)vr[c];
        }
        __bf16* d = sh_f1 + (size_t)row * SROW + c0;
#pragma unroll
        for (int c = 0; c < 32; ++c) d[c] = (__bf16)acc[c];
    }
    __syncthreads();

    // ---------- Phase 5: f2 = f1 @ Wo1^T + bo1 (16 col jobs, M=2, no relu) ----------
    for (int job = wave; job < 16; job += 8) {
        const __bf16* arows[2] = { sh_f1 + (size_t)(lane & 15) * SROW,
                                   sh_f1 + (size_t)(16 + (lane & 15)) * SROW };
        gemm_col<2, false>(Wo1B, pbo1, sh_f2, arows, job, lane);
    }
    __syncthreads();

    // ---------- Phase 6: out = f2 @ Wf^T + bf, store f32 [B,4,256] ----------
    for (int job = wave; job < 16; job += 8) {
        int ntile = job;
        const __bf16* arows[2] = { sh_f2 + (size_t)(lane & 15) * SROW,
                                   sh_f2 + (size_t)(16 + (lane & 15)) * SROW };
        v8f acc[2];
        { v8f z = {}; acc[0] = z; acc[1] = z; }
#pragma unroll
        for (int ks = 0; ks < 8; ++ks) {
            v16bf bfrag = load_b_frag(WfB, ntile, ks, lane);
#pragma unroll
            for (int m = 0; m < 2; ++m) {
                v16bf afrag = load_a_frag(arows[m], ks * 32, lane);
                acc[m] = __builtin_amdgcn_wmma_f32_16x16x32_bf16(false, afrag, false, bfrag,
                                                                 (short)0, acc[m], false, false);
            }
        }
        int n  = lane & 15;
        int mb = ((lane >> 4) & 1) * 8;
        float bb = pbf[ntile * 16 + n];
#pragma unroll
        for (int m = 0; m < 2; ++m) {
#pragma unroll
            for (int v = 0; v < 8; ++v) {
                int row = m * 16 + v + mb;              // 0..31 = b_local*4 + i
                int bl = row >> 2, i = row & 3;
                out[((size_t)(bbase + bl) * 4 + i) * DIM + ntile * 16 + n] = acc[m][v] + bb;
            }
        }
    }
}

extern "C" void kernel_launch(void* const* d_in, const int* in_sizes, int n_in,
                              void* d_out, int out_size, void* d_ws, size_t ws_size,
                              hipStream_t stream) {
    const float* h   = (const float*)d_in[0];
    const float* adj = (const float*)d_in[1];
    const float* lab = (const float*)d_in[2];
    const float* Wv  = (const float*)d_in[3];
    const float* bv  = (const float*)d_in[4];
    const float* Wk  = (const float*)d_in[5];
    const float* bk  = (const float*)d_in[6];
    const float* Wq  = (const float*)d_in[7];
    const float* bq  = (const float*)d_in[8];
    const float* Wo1 = (const float*)d_in[9];
    const float* bo1 = (const float*)d_in[10];
    const float* Wf  = (const float*)d_in[11];
    const float* bf  = (const float*)d_in[12];
    float* out = (float*)d_out;
    __bf16* wbf = (__bf16*)d_ws;                  // 5 * 65536 bf16 = 640 KB

    // pre-pass: fp32 -> bf16 weights into workspace (L2-resident afterwards)
    {
        int total_vec8 = 5 * WELEM / 8;           // 40960
        dim3 g(total_vec8 / 256), b(256);
        hipLaunchKernelGGL(cvt_weights_kernel, g, b, 0, stream, Wv, Wk, Wq, Wo1, Wf, wbf);
    }

    int batches = in_sizes[0] / (NODES * DIM);    // 8192
    dim3 grid(batches / GB);                      // 1024 workgroups
    dim3 block(256);                              // 8 waves (wave32)
    hipLaunchKernelGGL(gat_fused_kernel, grid, block, 0, stream,
                       h, adj, lab, wbf, bv, bk, bq, bo1, bf, out);
}
